// TransformerAttention_20461224198650
// MI455X (gfx1250) — matrix operand order repaired and checked by reading, not compiled, not run on hardware
//
#include <hip/hip_runtime.h>
#include <hip/hip_bf16.h>

typedef __attribute__((ext_vector_type(16))) __bf16 v16bf;
typedef __attribute__((ext_vector_type(8)))  __bf16 bf16x8;
typedef __attribute__((ext_vector_type(8)))  float  v8f;

#define B_SZ   2
#define T_SEQ  2048
#define NH     16
#define DH     64
#define DM     1024

#define ZERO8 {0.f,0.f,0.f,0.f,0.f,0.f,0.f,0.f}

// ---------------- WMMA helpers -----------------------------------------

__device__ __forceinline__ v8f wmma_bf16(v16bf a, v16bf b, v8f c) {
    // v_wmma_f32_16x16x32_bf16  (neg_a, A, neg_b, B, c_mod, C, reuse_a, reuse_b)
    return __builtin_amdgcn_wmma_f32_16x16x32_bf16(false, a, false, b, (short)0, c, false, false);
}

// A-fragment: 16(M) x 32(K) bf16, source row-major [rows, ld].
// Lane L: row = L&15 ; lanes 0-15 hold K=0..7 (v0-3) and 16..23 (v4-7),
// lanes 16-31 hold K=8..15 and 24..31  (ISA 7.12.2, 16-bit A layout).
__device__ __forceinline__ v16bf load_fragA(const __bf16* __restrict__ base, int ld,
                                            int row_base, int k_base, int lane) {
    int r    = lane & 15;
    int koff = (lane >> 4) << 3;          // 0 or 8
    const __bf16* p = base + (size_t)(row_base + r) * ld + k_base + koff;
    bf16x8 lo = *(const bf16x8*)p;        // K = koff .. koff+7
    bf16x8 hi = *(const bf16x8*)(p + 16); // K = koff+16 .. koff+23
    v16bf f;
#pragma unroll
    for (int i = 0; i < 8; ++i) { f[i] = lo[i]; f[i + 8] = hi[i]; }
    return f;
}

// B-fragment: 32(K) x 16(N) bf16, sourced from the *transposed* operand stored
// row-major [N, ld] (each lane's column is a contiguous row in memory).
// Lane L: col = L&15 ; lanes 0-15 hold K=0..7 and 16..23, lanes 16-31 hold K=8..15 and 24..31.
__device__ __forceinline__ v16bf load_fragB(const __bf16* __restrict__ base, int ld,
                                            int col_base, int k_base, int lane) {
    int n    = lane & 15;
    int koff = (lane >> 4) << 3;          // 0 or 8
    const __bf16* p = base + (size_t)(col_base + n) * ld + k_base + koff;
    bf16x8 lo = *(const bf16x8*)p;
    bf16x8 hi = *(const bf16x8*)(p + 16);
    v16bf f;
#pragma unroll
    for (int i = 0; i < 8; ++i) { f[i] = lo[i]; f[i + 8] = hi[i]; }
    return f;
}

// C/D tile (16x16 f32): lane L holds (M = 8*(L>>4)+j, N = L&15) in element j.
__device__ __forceinline__ void store_tile_f32(float* __restrict__ C, int ldc,
                                               int row_base, int col_base, int lane, v8f c) {
    int col = col_base + (lane & 15);
    int rb  = row_base + ((lane >> 4) << 3);
#pragma unroll
    for (int j = 0; j < 8; ++j) C[(size_t)(rb + j) * ldc + col] = c[j];
}

__device__ __forceinline__ float rowmax16(float v) {
#pragma unroll
    for (int m = 1; m < 16; m <<= 1) v = fmaxf(v, __shfl_xor(v, m, 32));
    return v;
}
__device__ __forceinline__ float rowsum16(float v) {
#pragma unroll
    for (int m = 1; m < 16; m <<= 1) v += __shfl_xor(v, m, 32);
    return v;
}

// ---------------- fp32 -> bf16 cast ------------------------------------

__global__ void cast_bf16_kernel(const float* __restrict__ src, __bf16* __restrict__ dst, int n) {
    int i = blockIdx.x * blockDim.x + threadIdx.x;
    if (i < n) dst[i] = (__bf16)src[i];
}

// ---------------- generic GEMM: C[M,N] = A[M,K] * B[N,K]^T --------------
// 256 threads = 8 waves tiled 4(M) x 2(N); wave tile 32x64 -> block 128x128.
// K-loop is software-pipelined: next-step fragments are loaded before the
// current-step WMMAs so matrix ops never wait on their own operands.
__global__ __launch_bounds__(256) void gemm_bf16_kernel(const __bf16* __restrict__ A,
                                                        const __bf16* __restrict__ Bm,
                                                        float* __restrict__ C,
                                                        int M, int N, int Kd) {
    int lane = threadIdx.x & 31;
    int wave = threadIdx.x >> 5;
    int wm   = wave & 3;
    int wn   = wave >> 2;
    int row0 = blockIdx.x * 128 + wm * 32;
    int col0 = blockIdx.y * 128 + wn * 64;

    v8f acc[2][4];
#pragma unroll
    for (int i = 0; i < 2; ++i)
#pragma unroll
        for (int j = 0; j < 4; ++j) acc[i][j] = (v8f)ZERO8;

    // prologue: fragments for k = 0
    v16bf a0 = load_fragA(A,  Kd, row0,      0, lane);
    v16bf a1 = load_fragA(A,  Kd, row0 + 16, 0, lane);
    v16bf b0 = load_fragB(Bm, Kd, col0,      0, lane);
    v16bf b1 = load_fragB(Bm, Kd, col0 + 16, 0, lane);
    v16bf b2 = load_fragB(Bm, Kd, col0 + 32, 0, lane);
    v16bf b3 = load_fragB(Bm, Kd, col0 + 48, 0, lane);

    for (int k = 32; k < Kd; k += 32) {
        if (k + 32 < Kd) {   // global_prefetch_b8 one step ahead on the A stream
            __builtin_prefetch(A + (size_t)(row0 + (lane & 15)) * Kd + k + 32, 0, 1);
        }
        // issue next-step loads first (independent registers)
        v16bf na0 = load_fragA(A,  Kd, row0,      k, lane);
        v16bf na1 = load_fragA(A,  Kd, row0 + 16, k, lane);
        v16bf nb0 = load_fragB(Bm, Kd, col0,      k, lane);
        v16bf nb1 = load_fragB(Bm, Kd, col0 + 16, k, lane);
        v16bf nb2 = load_fragB(Bm, Kd, col0 + 32, k, lane);
        v16bf nb3 = load_fragB(Bm, Kd, col0 + 48, k, lane);
        // 8 back-to-back WMMAs on already-resident fragments
        acc[0][0] = wmma_bf16(a0, b0, acc[0][0]);
        acc[0][1] = wmma_bf16(a0, b1, acc[0][1]);
        acc[0][2] = wmma_bf16(a0, b2, acc[0][2]);
        acc[0][3] = wmma_bf16(a0, b3, acc[0][3]);
        acc[1][0] = wmma_bf16(a1, b0, acc[1][0]);
        acc[1][1] = wmma_bf16(a1, b1, acc[1][1]);
        acc[1][2] = wmma_bf16(a1, b2, acc[1][2]);
        acc[1][3] = wmma_bf16(a1, b3, acc[1][3]);
        a0 = na0; a1 = na1; b0 = nb0; b1 = nb1; b2 = nb2; b3 = nb3;
    }
    // epilogue WMMAs for the last K step
    acc[0][0] = wmma_bf16(a0, b0, acc[0][0]);
    acc[0][1] = wmma_bf16(a0, b1, acc[0][1]);
    acc[0][2] = wmma_bf16(a0, b2, acc[0][2]);
    acc[0][3] = wmma_bf16(a0, b3, acc[0][3]);
    acc[1][0] = wmma_bf16(a1, b0, acc[1][0]);
    acc[1][1] = wmma_bf16(a1, b1, acc[1][1]);
    acc[1][2] = wmma_bf16(a1, b2, acc[1][2]);
    acc[1][3] = wmma_bf16(a1, b3, acc[1][3]);

#pragma unroll
    for (int i = 0; i < 2; ++i)
#pragma unroll
        for (int j = 0; j < 4; ++j)
            store_tile_f32(C, N, row0 + 16 * i, col0 + 16 * j, lane, acc[i][j]);
}

// ---------------- RoPE + scatter to Q/K [B,H,T,dh], V^T [B,H,dh,T] ------
// Softmax scale (1/sqrt(dh) = 0.125) folded into Q.
__global__ void rope_kernel(const float* __restrict__ qkv,
                            __bf16* __restrict__ Q, __bf16* __restrict__ K,
                            __bf16* __restrict__ Vt) {
    int idx = blockIdx.x * blockDim.x + threadIdx.x;   // b*T*NH + t*NH + h
    int h = idx & (NH - 1);
    int t = (idx >> 4) & (T_SEQ - 1);
    int b = idx >> 15;

    const float* row = qkv + (size_t)(b * T_SEQ + t) * (3 * DM);
    const float* q = row + h * DH;
    const float* k = row + DM + h * DH;
    const float* v = row + 2 * DM + h * DH;

    size_t obase = ((size_t)(b * NH + h) * T_SEQ + t) * DH;
    __bf16* qo = Q + obase;
    __bf16* ko = K + obase;
    __bf16* vo = Vt + (size_t)(b * NH + h) * DH * T_SEQ + t;

    const float log1e4_over_half = 9.210340372f / 32.0f;  // ln(10000)/(dh/2)
#pragma unroll
    for (int i = 0; i < 32; ++i) {
        float invf = __expf(-(float)i * log1e4_over_half);
        float ang  = (float)t * invf;
        float c = __cosf(ang), s = __sinf(ang);
        float q1 = q[i], q2 = q[i + 32];
        float k1 = k[i], k2 = k[i + 32];
        qo[i]      = (__bf16)((q1 * c - q2 * s) * 0.125f);
        qo[i + 32] = (__bf16)((q2 * c + q1 * s) * 0.125f);
        ko[i]      = (__bf16)(k1 * c - k2 * s);
        ko[i + 32] = (__bf16)(k2 * c + k1 * s);
        vo[(size_t)i * T_SEQ]        = (__bf16)v[i];
        vo[(size_t)(i + 32) * T_SEQ] = (__bf16)v[i + 32];
    }
}

// ---------------- causal flash attention --------------------------------
// grid = (T/128, B*H); 8 waves/block; each wave: 16 query rows, dh=64.
// Key blocks of 32; S via 4 WMMAs; V B-fragments are loaded BEFORE the
// softmax so global latency hides under the exp/shuffle VALU work; P is
// staged through a per-wave LDS tile (C-layout -> A-fragment transpose).
__global__ __launch_bounds__(256) void flash_attn_kernel(const __bf16* __restrict__ Q,
                                                         const __bf16* __restrict__ Km,
                                                         const __bf16* __restrict__ Vt,
                                                         __bf16* __restrict__ O) {
    __shared__ __align__(16) __bf16 plds[8 * 16 * 40];   // per-wave 16x32, stride 40

    int lane = threadIdx.x & 31;
    int wave = threadIdx.x >> 5;
    int bh   = blockIdx.y;                 // 0..B*NH-1
    int b    = bh >> 4;
    int h    = bh & (NH - 1);
    int q_base = blockIdx.x * 128 + wave * 16;

    const __bf16* Qbh = Q  + (size_t)bh * T_SEQ * DH;
    const __bf16* Kbh = Km + (size_t)bh * T_SEQ * DH;
    const __bf16* Vbh = Vt + (size_t)bh * DH * T_SEQ;

    v16bf aq0 = load_fragA(Qbh, DH, q_base, 0,  lane);
    v16bf aq1 = load_fragA(Qbh, DH, q_base, 32, lane);

    v8f o0 = ZERO8, o1 = ZERO8, o2 = ZERO8, o3 = ZERO8;
    float mrow[8], lrow[8];
#pragma unroll
    for (int j = 0; j < 8; ++j) { mrow[j] = -1.0e30f; lrow[j] = 0.0f; }

    __bf16* pw = plds + wave * 640;
    int rowM = (lane >> 4) << 3;
    int coln = lane & 15;

    int nkb = (q_base + 16 + 31) >> 5;     // 32-key blocks needed for causality

    for (int kb = 0; kb < nkb; ++kb) {
        int kstart = kb * 32;

        // ---- S = Q K^T (16 x 32), two 16x16 tiles ----
        v16bf bk0a = load_fragB(Kbh, DH, kstart,      0,  lane);
        v16bf bk0b = load_fragB(Kbh, DH, kstart,      32, lane);
        v16bf bk1a = load_fragB(Kbh, DH, kstart + 16, 0,  lane);
        v16bf bk1b = load_fragB(Kbh, DH, kstart + 16, 32, lane);
        v8f s0 = ZERO8, s1 = ZERO8;
        s0 = wmma_bf16(aq0, bk0a, s0);
        s0 = wmma_bf16(aq1, bk0b, s0);
        s1 = wmma_bf16(aq0, bk1a, s1);
        s1 = wmma_bf16(aq1, bk1b, s1);

        // ---- V fragments issued early: latency hidden by softmax VALU ----
        v16bf bv0 = load_fragB(Vbh, T_SEQ, 0,  kstart, lane);
        v16bf bv1 = load_fragB(Vbh, T_SEQ, 16, kstart, lane);
        v16bf bv2 = load_fragB(Vbh, T_SEQ, 32, kstart, lane);
        v16bf bv3 = load_fragB(Vbh, T_SEQ, 48, kstart, lane);

        // ---- online softmax (per-row stats via 16-lane butterflies) ----
#pragma unroll
        for (int j = 0; j < 8; ++j) {
            int qrow = q_base + rowM + j;
            int k0 = kstart + coln;
            float v0 = (k0      <= qrow) ? s0[j] : -3.0e38f;
            float v1 = (k0 + 16 <= qrow) ? s1[j] : -3.0e38f;
            float mnew  = fmaxf(mrow[j], rowmax16(fmaxf(v0, v1)));
            float alpha = __expf(mrow[j] - mnew);
            float p0 = __expf(v0 - mnew);
            float p1 = __expf(v1 - mnew);
            lrow[j] = lrow[j] * alpha + rowsum16(p0 + p1);
            mrow[j] = mnew;
            o0[j] *= alpha; o1[j] *= alpha; o2[j] *= alpha; o3[j] *= alpha;
            pw[(rowM + j) * 40 + coln]      = (__bf16)p0;
            pw[(rowM + j) * 40 + coln + 16] = (__bf16)p1;
        }
        // drain LDS stores before re-reading as A-fragment (same wave, DScnt)
        asm volatile("s_wait_dscnt 0x0" ::: "memory");

        // ---- O += P V ----
        v16bf ap = load_fragA(pw, 40, 0, 0, lane);
        o0 = wmma_bf16(ap, bv0, o0);
        o1 = wmma_bf16(ap, bv1, o1);
        o2 = wmma_bf16(ap, bv2, o2);
        o3 = wmma_bf16(ap, bv3, o3);
    }

    // ---- normalize and store to [B, T, H*dh] (bf16 for the final GEMM) ----
#pragma unroll
    for (int j = 0; j < 8; ++j) {
        float inv = 1.0f / lrow[j];
        int row = q_base + rowM + j;
        size_t base = ((size_t)(b * T_SEQ + row)) * DM + h * DH + coln;
        O[base + 0]  = (__bf16)(o0[j] * inv);
        O[base + 16] = (__bf16)(o1[j] * inv);
        O[base + 32] = (__bf16)(o2[j] * inv);
        O[base + 48] = (__bf16)(o3[j] * inv);
    }
}

// ---------------- launch ------------------------------------------------

extern "C" void kernel_launch(void* const* d_in, const int* in_sizes, int n_in,
                              void* d_out, int out_size, void* d_ws, size_t ws_size,
                              hipStream_t stream) {
    const float* x     = (const float*)d_in[0];   // [B,T,DM]
    const float* qkv_w = (const float*)d_in[1];   // [3*DM, DM]
    const float* out_w = (const float*)d_in[2];   // [DM, DM]
    float* out = (float*)d_out;                   // [B,T,DM]

    const size_t MROWS = (size_t)B_SZ * T_SEQ;    // 4096

    char* ws = (char*)d_ws;
    size_t off = 0;
    auto alloc = [&](size_t bytes) -> void* {
        void* p = ws + off;
        off += (bytes + 255) & ~(size_t)255;
        return p;
    };
    __bf16* xbf  = (__bf16*)alloc(MROWS * DM * 2);                     // 8 MB
    __bf16* wqkv = (__bf16*)alloc((size_t)3 * DM * DM * 2);            // 6 MB
    __bf16* wout = (__bf16*)alloc((size_t)DM * DM * 2);                // 2 MB
    float*  qkv  = (float*) alloc(MROWS * 3 * DM * 4);                 // 48 MB
    __bf16* Qb   = (__bf16*)alloc((size_t)B_SZ * NH * T_SEQ * DH * 2); // 8 MB
    __bf16* Kb   = (__bf16*)alloc((size_t)B_SZ * NH * T_SEQ * DH * 2); // 8 MB
    __bf16* Vt   = (__bf16*)alloc((size_t)B_SZ * NH * DH * T_SEQ * 2); // 8 MB
    __bf16* Ob   = (__bf16*)alloc(MROWS * DM * 2);                     // 8 MB

    int n_x = (int)(MROWS * DM);
    int n_w = 3 * DM * DM;
    int n_o = DM * DM;
    cast_bf16_kernel<<<(n_x + 255) / 256, 256, 0, stream>>>(x, xbf, n_x);
    cast_bf16_kernel<<<(n_w + 255) / 256, 256, 0, stream>>>(qkv_w, wqkv, n_w);
    cast_bf16_kernel<<<(n_o + 255) / 256, 256, 0, stream>>>(out_w, wout, n_o);

    // qkv[4096, 3072] = x[4096,1024] * qkv_w[3072,1024]^T
    gemm_bf16_kernel<<<dim3(MROWS / 128, (3 * DM) / 128), 256, 0, stream>>>(
        xbf, wqkv, qkv, (int)MROWS, 3 * DM, DM);

    // RoPE + scatter to Q/K [B,H,T,dh] and V^T [B,H,dh,T]
    int n_rope = B_SZ * T_SEQ * NH;
    rope_kernel<<<n_rope / 256, 256, 0, stream>>>(qkv, Qb, Kb, Vt);

    // causal flash attention -> Ob [B,T,DM] bf16
    flash_attn_kernel<<<dim3(T_SEQ / 128, B_SZ * NH), 256, 0, stream>>>(Qb, Kb, Vt, Ob);

    // out[4096,1024] = Ob[4096,1024] * out_w[1024,1024]^T
    gemm_bf16_kernel<<<dim3(MROWS / 128, DM / 128), 256, 0, stream>>>(
        Ob, wout, out, (int)MROWS, DM, DM);
}
